// RSSM_2972117369079
// MI455X (gfx1250) — matrix operand order repaired and checked by reading, not compiled, not run on hardware
//
#include <hip/hip_runtime.h>

#define T_  64
#define B_  256
#define S_  1024
#define A_  32
#define H_  2048
#define KX_ 1056           // S_+A_
#define KTX 33             // KX_/32 k-tiles
#define KTH 64             // H_/32 k-tiles

typedef __attribute__((ext_vector_type(16))) __bf16 bf16x16;
typedef __attribute__((ext_vector_type(8)))  float  f32x8;

// ---- f32 -> bf16 round-to-nearest-even ----
__device__ __forceinline__ unsigned short f2bf(float f) {
  union { float f; unsigned u; } v; v.f = f;
  unsigned r = v.u + 0x7FFFu + ((v.u >> 16) & 1u);
  return (unsigned short)(r >> 16);
}

// A-matrix (16x32 bf16) fragment slot for element (m, k), in ushort units.
// Layout per CDNA5 ISA 7.12.2: lanes 0-15 hold K={0..7,16..23}, lanes 16-31 K={8..15,24..31},
// VGPR j holds a packed K pair.
__device__ __forceinline__ size_t packOff(int m, int k) {
  int kt = k >> 5, kk = k & 31;
  int ln = m + (((kk >> 3) & 1) << 4);
  int j  = ((kk & 7) >> 1) + (((kk >> 4) & 1) << 2);
  return (size_t)kt * 512 + (size_t)(ln * 16 + j * 2 + (kk & 1));
}

__device__ __forceinline__ f32x8 wmma_bf(const unsigned short* a,
                                         const unsigned short* b, f32x8 c) {
  bf16x16 av = *(const bf16x16*)a;
  bf16x16 bv = *(const bf16x16*)b;
  return __builtin_amdgcn_wmma_f32_16x16x32_bf16(false, av, false, bv,
                                                 (short)0, c, false, false);
}

// CDNA5 async copy global -> LDS (GLOBAL_LOAD_ASYNC_TO_LDS_B128, ASYNCcnt-tracked).
// One instruction per wave moves 32 lanes x 16B = 512B. bytes must be multiple of 16.
__device__ __forceinline__ void asyncStage(void* ldsDst, const void* gSrc, int bytes) {
  // flat pointers to LDS keep the wave-relative LDS byte offset in addr[31:0]
  unsigned lbase = (unsigned)(uintptr_t)ldsDst;
  const char* g = (const char*)gSrc;
  for (int i = threadIdx.x * 16; i < bytes; i += 256 * 16) {
    unsigned la = lbase + (unsigned)i;
    const void* ga = g + i;
    asm volatile("global_load_async_to_lds_b128 %0, %1, off"
                 :: "v"(la), "v"(ga) : "memory");
  }
  asm volatile("s_wait_asynccnt 0x0" ::: "memory");
}

// inter-workgroup (16 WGs of one batch-row group) barrier via L2 atomics
__device__ __forceinline__ void groupBarrier(int* cnt) {
  __threadfence();
  __syncthreads();
  if (threadIdx.x == 0) {
    __hip_atomic_fetch_add(cnt, 1, __ATOMIC_RELEASE, __HIP_MEMORY_SCOPE_AGENT);
    while (__hip_atomic_load(cnt, __ATOMIC_ACQUIRE, __HIP_MEMORY_SCOPE_AGENT) < 16)
      __builtin_amdgcn_s_sleep(1);
  }
  __syncthreads();
  __threadfence();
}

// ---------------- pre-pass kernels ----------------
__global__ void zeroCnt(int* c, int n) {
  for (int i = blockIdx.x * blockDim.x + threadIdx.x; i < n;
       i += blockDim.x * gridDim.x) c[i] = 0;
}

// Pack a weight matrix into B-fragment-major bf16 layout.
// B 32x16 tile layout: lanes 0-15 hold K={0..7,16..23}, lanes 16-31 K={8..15,24..31}; VGPR j holds a packed K pair.
// srcNK==0: src is [K,N] row-major (W_tp, W_post). srcNK==1: src is [N,K] (W_ih, W_hh, B=src^T).
__global__ void packW(const float* __restrict__ src, unsigned short* __restrict__ dst,
                      int K, int N, int srcNK) {
  int KT = K >> 5;
  size_t total = (size_t)K * N;
  for (size_t idx = (size_t)blockIdx.x * blockDim.x + threadIdx.x; idx < total;
       idx += (size_t)gridDim.x * blockDim.x) {
    int n = (int)(idx % N);
    int k = (int)(idx / N);
    float v = srcNK ? src[(size_t)n * K + k] : src[(size_t)k * N + n];
    int kk = k & 31;
    int ln = (n & 15) + (((kk >> 3) & 1) << 4);
    int j  = ((kk & 7) >> 1) + (((kk >> 4) & 1) << 2);
    size_t frag = (size_t)(n >> 4) * KT + (k >> 5);
    dst[frag * 512 + ln * 16 + j * 2 + (kk & 1)] = f2bf(v);
  }
}

// ---------------- main persistent RSSM kernel ----------------
// grid: 256 WGs = 16 batch-row groups (g) x 16 N-slices (s); block: 256 thr = 8 wave32.
// dynamic LDS carve: x tiles (33KB) | hidden tiles (64KB) | std exchange (4KB)
__global__ void __launch_bounds__(256) rssm_kernel(
    const float* __restrict__ prev_state, const float* __restrict__ actions,
    const float* __restrict__ prev_belief, const float* __restrict__ eps,
    const float* __restrict__ b_ih, const float* __restrict__ b_hh,
    const float* __restrict__ b_tp, const float* __restrict__ b_post,
    const unsigned short* __restrict__ wTp, const unsigned short* __restrict__ wIh,
    const unsigned short* __restrict__ wHh, const unsigned short* __restrict__ wPo,
    unsigned short* __restrict__ xPack, unsigned short* __restrict__ belPack,
    unsigned short* __restrict__ hidPack, int* __restrict__ counters,
    float* __restrict__ outB, float* __restrict__ outS,
    float* __restrict__ outM, float* __restrict__ outD) {
  extern __shared__ __align__(16) unsigned short dynLds[];
  unsigned short* xLds   = dynLds;                    // KTX*512 ushorts = 33792 B
  unsigned short* hidLds = dynLds + KTX * 512;        // KTH*512 ushorts = 65536 B
  float* stdLds = (float*)(dynLds + (KTX + KTH) * 512);  // 1024 floats = 4096 B

  const int tid  = threadIdx.x;
  const int wave = tid >> 5;
  const int lane = tid & 31;
  const int lh   = lane >> 4;   // which 16-lane half
  const int lc   = lane & 15;   // N column within tile
  const int g    = blockIdx.x >> 4;
  const int s    = blockIdx.x & 15;
  const int m0   = g * 16;

  unsigned short* hidG = hidPack + (size_t)g * (KTH * 512);
  int* cbase = counters + g * 256;
  int barIdx = 0;
  int par = 0;

  // ---- prologue: pack x(t=0)=[state,action] and belief(0); init belief regs ----
  {
    unsigned short* xp0 = xPack + (size_t)g * (KTX * 512);
    unsigned short* bp0 = belPack + (size_t)g * (KTH * 512);
    for (int e = s * 256 + tid; e < 16 * KX_; e += 4096) {
      int m = e & 15, k = e >> 4;
      float v = (k < S_) ? prev_state[(size_t)(m0 + m) * S_ + k]
                         : actions[(size_t)(m0 + m) * A_ + (k - S_)];
      xp0[packOff(m, k)] = f2bf(v);
    }
    for (int e = s * 256 + tid; e < 16 * H_; e += 4096) {
      int m = e & 15, k = e >> 4;
      bp0[packOff(m, k)] = f2bf(prev_belief[(size_t)(m0 + m) * H_ + k]);
    }
  }
  f32x8 bel;
  {
    int col = s * 128 + wave * 16 + lc;
#pragma unroll
    for (int i = 0; i < 8; ++i)
      bel[i] = prev_belief[(size_t)(m0 + i + 8 * lh) * H_ + col];
  }
  groupBarrier(cbase + (barIdx++));

  for (int t = 0; t < T_; ++t) {
    const unsigned short* xpCur = xPack + (size_t)(par * 16 + g) * (KTX * 512);
    unsigned short* xpNxt = xPack + (size_t)((par ^ 1) * 16 + g) * (KTX * 512);
    const unsigned short* bpCur = belPack + (size_t)(par * 16 + g) * (KTH * 512);
    unsigned short* bpNxt = belPack + (size_t)((par ^ 1) * 16 + g) * (KTH * 512);

    // phase 0: async-stage packed x into LDS (shared by all 8 waves)
    asyncStage(xLds, xpCur, KTX * 512 * 2);
    __syncthreads();

    // phase 1: hidden = relu(x @ W_tp + b_tp), 16x16 tile per wave, K=1056
    {
      f32x8 acc = {0.f, 0.f, 0.f, 0.f, 0.f, 0.f, 0.f, 0.f};
      const unsigned short* pB = wTp + (size_t)(s * 8 + wave) * (KTX * 512) + lane * 16;
#pragma unroll 3
      for (int kt = 0; kt < KTX; ++kt) {
        acc = wmma_bf(&xLds[kt * 512 + lane * 16], pB, acc);
        pB += 512;
      }
      int nt = s * 128 + wave * 16;
      float btp = b_tp[nt + lc];
#pragma unroll
      for (int i = 0; i < 8; ++i) {
        float h = fmaxf(acc[i] + btp, 0.f);
        hidG[packOff(i + 8 * lh, nt + lc)] = f2bf(h);
      }
    }
    groupBarrier(cbase + (barIdx++));  // hidden visible to whole group

    // stage full packed hidden (64KB) into LDS via async engine; used by phases 2+3
    asyncStage(hidLds, hidG, KTH * 512 * 2);
    __syncthreads();

    // phase 2: GRU gates -> new belief (6 accumulators per wave, K=2048)
    {
      f32x8 air = {0.f,0.f,0.f,0.f,0.f,0.f,0.f,0.f};
      f32x8 aiz = air, ain = air, ahr = air, ahz = air, ahn = air;
      const unsigned short* pAb = bpCur + lane * 16;
      int ct = s * 8 + wave;
      const unsigned short* pIr = wIh + (size_t)(0   + ct) * (KTH * 512) + lane * 16;
      const unsigned short* pIz = wIh + (size_t)(128 + ct) * (KTH * 512) + lane * 16;
      const unsigned short* pIn = wIh + (size_t)(256 + ct) * (KTH * 512) + lane * 16;
      const unsigned short* pHr = wHh + (size_t)(0   + ct) * (KTH * 512) + lane * 16;
      const unsigned short* pHz = wHh + (size_t)(128 + ct) * (KTH * 512) + lane * 16;
      const unsigned short* pHn = wHh + (size_t)(256 + ct) * (KTH * 512) + lane * 16;
#pragma unroll 2
      for (int kt = 0; kt < KTH; ++kt) {
        bf16x16 ah = *(const bf16x16*)&hidLds[kt * 512 + lane * 16];
        bf16x16 ab = *(const bf16x16*)pAb;
        air = __builtin_amdgcn_wmma_f32_16x16x32_bf16(false, ah, false, *(const bf16x16*)pIr, (short)0, air, false, false);
        aiz = __builtin_amdgcn_wmma_f32_16x16x32_bf16(false, ah, false, *(const bf16x16*)pIz, (short)0, aiz, false, false);
        ain = __builtin_amdgcn_wmma_f32_16x16x32_bf16(false, ah, false, *(const bf16x16*)pIn, (short)0, ain, false, false);
        ahr = __builtin_amdgcn_wmma_f32_16x16x32_bf16(false, ab, false, *(const bf16x16*)pHr, (short)0, ahr, false, false);
        ahz = __builtin_amdgcn_wmma_f32_16x16x32_bf16(false, ab, false, *(const bf16x16*)pHz, (short)0, ahz, false, false);
        ahn = __builtin_amdgcn_wmma_f32_16x16x32_bf16(false, ab, false, *(const bf16x16*)pHn, (short)0, ahn, false, false);
        pAb += 512; pIr += 512; pIz += 512; pIn += 512;
        pHr += 512; pHz += 512; pHn += 512;
      }
      int col = s * 128 + wave * 16 + lc;
      float bir = b_ih[col], biz = b_ih[H_ + col], bin = b_ih[2 * H_ + col];
      float bhr = b_hh[col], bhz = b_hh[H_ + col], bhn = b_hh[2 * H_ + col];
#pragma unroll
      for (int i = 0; i < 8; ++i) {
        float r  = 1.f / (1.f + expf(-(air[i] + ahr[i] + bir + bhr)));
        float z  = 1.f / (1.f + expf(-(aiz[i] + ahz[i] + biz + bhz)));
        float nn = tanhf(ain[i] + bin + r * (ahn[i] + bhn));
        float nb = (1.f - z) * nn + z * bel[i];
        bel[i] = nb;
        int m = m0 + i + 8 * lh;
        outB[((size_t)t * B_ + m) * H_ + col] = nb;
        bpNxt[packOff(i + 8 * lh, col)] = f2bf(nb);
      }
    }

    // phase 3: posterior stats: waves 0-3 -> mean tiles, waves 4-7 -> std tiles
    {
      int isStd = (wave >= 4);
      int w4 = wave & 3;
      int ntile = (isStd ? 64 : 0) + s * 4 + w4;
      f32x8 acc = {0.f, 0.f, 0.f, 0.f, 0.f, 0.f, 0.f, 0.f};
      const unsigned short* pB = wPo + (size_t)ntile * (KTH * 512) + lane * 16;
#pragma unroll 4
      for (int kt = 0; kt < KTH; ++kt) {
        acc = wmma_bf(&hidLds[kt * 512 + lane * 16], pB, acc);
        pB += 512;
      }
      int colP = (isStd ? 1024 : 0) + s * 64 + w4 * 16 + lc;
      float bp = b_post[colP];
      if (isStd) {
#pragma unroll
        for (int i = 0; i < 8; ++i) {
          float x = acc[i] + bp;
          float sd = (x > 20.f) ? x : log1pf(expf(x));  // softplus
          int mr = i + 8 * lh;
          outD[((size_t)t * B_ + m0 + mr) * S_ + (colP - 1024)] = sd;
          stdLds[w4 * 256 + mr * 16 + lc] = sd;
        }
      }
      __syncthreads();
      if (!isStd) {
#pragma unroll
        for (int i = 0; i < 8; ++i) {
          int mr = i + 8 * lh;
          float mn = acc[i] + bp;
          float sd = stdLds[w4 * 256 + mr * 16 + lc];
          float ep = eps[((size_t)t * B_ + m0 + mr) * S_ + colP];
          float st = mn + sd * ep;
          outM[((size_t)t * B_ + m0 + mr) * S_ + colP] = mn;
          outS[((size_t)t * B_ + m0 + mr) * S_ + colP] = st;
          xpNxt[packOff(mr, colP)] = f2bf(st);  // state feeds next step's x
        }
      }
      if (s == 0 && t + 1 < T_) {  // pack next action tile (k-tile 32, disjoint)
        for (int e = tid; e < 512; e += 256) {
          int m = e >> 5, a = e & 31;
          float v = actions[((size_t)(t + 1) * B_ + m0 + m) * A_ + a];
          xpNxt[packOff(m, S_ + a)] = f2bf(v);
        }
      }
    }
    groupBarrier(cbase + (barIdx++));
    par ^= 1;
  }
}

extern "C" void kernel_launch(void* const* d_in, const int* in_sizes, int n_in,
                              void* d_out, int out_size, void* d_ws, size_t ws_size,
                              hipStream_t stream) {
  (void)in_sizes; (void)n_in; (void)out_size; (void)ws_size;
  const float* prev_state  = (const float*)d_in[0];
  const float* actions     = (const float*)d_in[1];
  const float* prev_belief = (const float*)d_in[2];
  const float* eps         = (const float*)d_in[3];
  const float* W_ih = (const float*)d_in[4];
  const float* W_hh = (const float*)d_in[5];
  const float* b_ih = (const float*)d_in[6];
  const float* b_hh = (const float*)d_in[7];
  const float* W_tp = (const float*)d_in[8];
  const float* b_tp = (const float*)d_in[9];
  const float* W_post = (const float*)d_in[10];
  const float* b_post = (const float*)d_in[11];

  char* ws = (char*)d_ws;
  size_t off = 0;
  int* counters = (int*)(ws + off);              off += (size_t)16 * 256 * 4;
  unsigned short* wTpP = (unsigned short*)(ws + off); off += (size_t)128 * 33 * 1024;
  unsigned short* wIhP = (unsigned short*)(ws + off); off += (size_t)384 * 64 * 1024;
  unsigned short* wHhP = (unsigned short*)(ws + off); off += (size_t)384 * 64 * 1024;
  unsigned short* wPoP = (unsigned short*)(ws + off); off += (size_t)128 * 64 * 1024;
  unsigned short* xP   = (unsigned short*)(ws + off); off += (size_t)2 * 16 * 33 * 1024;
  unsigned short* bP   = (unsigned short*)(ws + off); off += (size_t)2 * 16 * 64 * 1024;
  unsigned short* hP   = (unsigned short*)(ws + off); off += (size_t)16 * 64 * 1024;

  float* outB = (float*)d_out;
  float* outS = outB + (size_t)T_ * B_ * H_;
  float* outM = outS + (size_t)T_ * B_ * S_;
  float* outD = outM + (size_t)T_ * B_ * S_;

  const int ldsBytes = (KTX + KTH) * 1024 + 4096;  // 103424 B dynamic LDS
  (void)hipFuncSetAttribute((const void*)rssm_kernel,
                            hipFuncAttributeMaxDynamicSharedMemorySize, ldsBytes);

  zeroCnt<<<4, 256, 0, stream>>>(counters, 16 * 256);
  packW<<<2048, 256, 0, stream>>>(W_tp,   wTpP, KX_, H_,     0);
  packW<<<4096, 256, 0, stream>>>(W_ih,   wIhP, H_,  3 * H_, 1);
  packW<<<4096, 256, 0, stream>>>(W_hh,   wHhP, H_,  3 * H_, 1);
  packW<<<2048, 256, 0, stream>>>(W_post, wPoP, H_,  2 * S_, 0);
  rssm_kernel<<<256, 256, ldsBytes, stream>>>(prev_state, actions, prev_belief, eps,
      b_ih, b_hh, b_tp, b_post, wTpP, wIhP, wHhP, wPoP, xP, bP, hP, counters,
      outB, outS, outM, outD);
}
